// topk_PAM_Module_12807592476758
// MI455X (gfx1250) — matrix operand order repaired and checked by reading, not compiled, not run on hardware
//
#include <hip/hip_runtime.h>
#include <hip/hip_bf16.h>

// topk-PAM attention for MI455X (gfx1250, wave32, WMMA).
// Heavy matmuls use v_wmma_f32_16x16x32_bf16 (bf16 in, f32 accumulate) with
// 2x2 register blocking per wave (32x32 output region, 4 accumulators) to
// halve fragment loads per WMMA and hide WMMA->VALU hazard latency.
// The B x N x N energy/attention matrix is kept in bf16 (128 MB) so it fits
// the 192 MB L2 and halves HBM traffic vs f32 at 23.3 TB/s.
//
// Workspace layout (needs ~165 MB of d_ws):
//   xT  : B*N*C  bf16 (x transposed so GEMM K-runs are contiguous)
//   qT  : B*N*KD bf16
//   kT  : B*N*KD bf16
//   vbf : B*C*N  bf16
//   e   : B*N*N  bf16 (energy, overwritten in-place with attn probs)

typedef __attribute__((ext_vector_type(8)))  float  v8f;
typedef __attribute__((ext_vector_type(8)))  float  f32x8;
typedef __attribute__((ext_vector_type(8)))  __bf16 bf16x8;
typedef __attribute__((ext_vector_type(16))) __bf16 bf16x16;

constexpr int BATCH = 4;
constexpr int CH    = 512;
constexpr int KDIM  = 64;
constexpr int NPIX  = 4096;

// ---------------- WMMA fragment loaders (ISA 7.12.2 layouts, wave32) -------
// A-matrix 16x32 bf16: lane (half=lane>>4, r=lane&15) holds row M=r,
//   elements 0..7  = K = half*8 + 0..7
//   elements 8..15 = K = 16 + half*8 + 0..7      -> two aligned 16B loads.
__device__ __forceinline__ bf16x16 frag_a_bf16(const __bf16* __restrict__ base,
                                               int ld, int row, int k, int lane) {
  const int half = lane >> 4, r = lane & 15;
  const __bf16* p = base + (size_t)(row + r) * ld + k + half * 8;
  bf16x8 lo = *(const bf16x8*)(p);
  bf16x8 hi = *(const bf16x8*)(p + 16);
  return __builtin_shufflevector(lo, hi, 0,1,2,3,4,5,6,7,8,9,10,11,12,13,14,15);
}

__device__ __forceinline__ bf16x16 frag_a_f32(const float* __restrict__ base,
                                              int ld, int row, int k, int lane) {
  const int half = lane >> 4, r = lane & 15;
  const float* p = base + (size_t)(row + r) * ld + k + half * 8;
  f32x8 lo = *(const f32x8*)(p);
  f32x8 hi = *(const f32x8*)(p + 16);
  bf16x16 out;
#pragma unroll
  for (int i = 0; i < 8; ++i) { out[i] = (__bf16)lo[i]; out[i + 8] = (__bf16)hi[i]; }
  return out;
}

// B-matrix 32x16 bf16: lane holds column N=r, K = half*8 + 0..7 and
//   16 + half*8 + 0..7 -> two aligned 16B loads per lane.
__device__ __forceinline__ bf16x16 frag_b_bf16(const __bf16* __restrict__ base,
                                               int ld, int row, int k, int lane) {
  const int half = lane >> 4, r = lane & 15;
  const __bf16* p = base + (size_t)(row + r) * ld + k + half * 8;
  bf16x8 lo = *(const bf16x8*)(p);
  bf16x8 hi = *(const bf16x8*)(p + 16);
  return __builtin_shufflevector(lo, hi, 0,1,2,3,4,5,6,7,8,9,10,11,12,13,14,15);
}

__device__ __forceinline__ bf16x16 frag_b_f32(const float* __restrict__ base,
                                              int ld, int row, int k, int lane) {
  const int half = lane >> 4, r = lane & 15;
  const float* p = base + (size_t)(row + r) * ld + k + half * 8;
  f32x8 lo = *(const f32x8*)(p);
  f32x8 hi = *(const f32x8*)(p + 16);
  bf16x16 out;
#pragma unroll
  for (int i = 0; i < 8; ++i) { out[i] = (__bf16)lo[i]; out[i + 8] = (__bf16)hi[i]; }
  return out;
}

#define WMMA_BF16(A, Bm, Cacc) \
  __builtin_amdgcn_wmma_f32_16x16x32_bf16(false, (A), false, (Bm), (short)0, (Cacc), false, false)

// ---------------- Stage 0: x (B,C,N) f32 -> xT (B,N,C) bf16 ----------------
__global__ __launch_bounds__(1024) void k_transpose(const float* __restrict__ x,
                                                    __bf16* __restrict__ xT) {
  __shared__ float tile[32][33];
  const int b  = blockIdx.z;
  const float* xb = x + (size_t)b * CH * NPIX;
  __bf16* xTb = xT + (size_t)b * NPIX * CH;
  const int n0 = blockIdx.x * 32, c0 = blockIdx.y * 32;
  tile[threadIdx.y][threadIdx.x] =
      xb[(size_t)(c0 + threadIdx.y) * NPIX + n0 + threadIdx.x];
  __syncthreads();
  xTb[(size_t)(n0 + threadIdx.y) * CH + c0 + threadIdx.x] =
      (__bf16)tile[threadIdx.x][threadIdx.y];
}

// ---------------- Stage 1a: qT/kT (N x KD bf16) ----------------------------
// D[m = proj-row][n = pixel]; transposed store -> contiguous bf16x8 per lane.
__global__ __launch_bounds__(256) void k_proj_qk(const float* __restrict__ W,
                                                 const float* __restrict__ bias,
                                                 const __bf16* __restrict__ xT,
                                                 __bf16* __restrict__ outT) {
  const int lane = threadIdx.x & 31;
  const int tile = blockIdx.x * 8 + (threadIdx.x >> 5);
  const int TM = KDIM / 16, TN = NPIX / 16;
  const int b = tile / (TM * TN);
  const int t = tile % (TM * TN);
  const int tm = t / TN, tn = t % TN;
  const __bf16* xTb = xT + (size_t)b * NPIX * CH;
  __bf16* outTb = outT + (size_t)b * NPIX * KDIM;
  v8f acc = {};
  for (int k = 0; k < CH; k += 32) {
    bf16x16 a  = frag_a_f32(W, CH, tm * 16, k, lane);
    bf16x16 bm = frag_b_bf16(xTb, CH, tn * 16, k, lane);
    acc = WMMA_BF16(a, bm, acc);
  }
  const int half = lane >> 4, r = lane & 15;
  const int mb = tm * 16 + half * 8;
  const int n  = tn * 16 + r;
  bf16x8 o;
#pragma unroll
  for (int v = 0; v < 8; ++v) o[v] = (__bf16)(acc[v] + bias[mb + v]);
  *(bf16x8*)(outTb + (size_t)n * KDIM + mb) = o;
}

// ---------------- Stage 1b: v (C x N bf16) ---------------------------------
// D[m = pixel][n = channel] so stores along the pixel dim vectorize.
__global__ __launch_bounds__(256) void k_proj_v(const float* __restrict__ Wv,
                                                const float* __restrict__ bv,
                                                const __bf16* __restrict__ xT,
                                                __bf16* __restrict__ vbf) {
  const int lane = threadIdx.x & 31;
  const int tile = blockIdx.x * 8 + (threadIdx.x >> 5);
  const int TN = NPIX / 16, TC = CH / 16;
  const int b = tile / (TN * TC);
  const int t = tile % (TN * TC);
  const int tn = t / TC, tc = t % TC;
  const __bf16* xTb = xT + (size_t)b * NPIX * CH;
  __bf16* vb = vbf + (size_t)b * CH * NPIX;
  v8f acc = {};
  for (int k = 0; k < CH; k += 32) {
    bf16x16 a  = frag_a_bf16(xTb, CH, tn * 16, k, lane);
    bf16x16 bm = frag_b_f32(Wv, CH, tc * 16, k, lane);
    acc = WMMA_BF16(a, bm, acc);
  }
  const int half = lane >> 4, r = lane & 15;
  const int c  = tc * 16 + r;
  const int nb = tn * 16 + half * 8;
  const float bias = bv[c];
  bf16x8 o;
#pragma unroll
  for (int v = 0; v < 8; ++v) o[v] = (__bf16)(acc[v] + bias);
  *(bf16x8*)(vb + (size_t)c * NPIX + nb) = o;
}

// ---------------- Stage 2: energy[i][j] = q(:,i).k(:,j), bf16 --------------
// 2x2 register-blocked: each wave computes a 32x32 region as D[m=j][n=i]
// tiles so the row-major bf16 stores vectorize. 4 frag loads feed 4 WMMAs.
__global__ __launch_bounds__(256) void k_energy(const __bf16* __restrict__ qT,
                                                const __bf16* __restrict__ kT,
                                                __bf16* __restrict__ e) {
  const int lane = threadIdx.x & 31;
  const int tile = blockIdx.x * 8 + (threadIdx.x >> 5);
  const int TI2 = NPIX / 32, TJ2 = NPIX / 32;
  const int b = tile / (TI2 * TJ2);
  const int t = tile % (TI2 * TJ2);
  const int ti2 = t / TJ2, tj2 = t % TJ2;
  const __bf16* qb = qT + (size_t)b * NPIX * KDIM;
  const __bf16* kb = kT + (size_t)b * NPIX * KDIM;
  __bf16* eb = e + (size_t)b * NPIX * NPIX;
  v8f acc00 = {}, acc01 = {}, acc10 = {}, acc11 = {};
#pragma unroll
  for (int k = 0; k < KDIM; k += 32) {
    bf16x16 a0 = frag_a_bf16(kb, KDIM, tj2 * 32,      k, lane);  // M = j
    bf16x16 a1 = frag_a_bf16(kb, KDIM, tj2 * 32 + 16, k, lane);
    bf16x16 b0 = frag_b_bf16(qb, KDIM, ti2 * 32,      k, lane);  // N = i
    bf16x16 b1 = frag_b_bf16(qb, KDIM, ti2 * 32 + 16, k, lane);
    acc00 = WMMA_BF16(a0, b0, acc00);
    acc10 = WMMA_BF16(a1, b0, acc10);
    acc01 = WMMA_BF16(a0, b1, acc01);
    acc11 = WMMA_BF16(a1, b1, acc11);
  }
  const int half = lane >> 4, r = lane & 15;
  const v8f accs[2][2] = {{acc00, acc01}, {acc10, acc11}};
#pragma unroll
  for (int jt = 0; jt < 2; ++jt) {
#pragma unroll
    for (int it = 0; it < 2; ++it) {
      const int i  = ti2 * 32 + it * 16 + r;
      const int jb = tj2 * 32 + jt * 16 + half * 8;
      bf16x8 o;
#pragma unroll
      for (int v = 0; v < 8; ++v) o[v] = (__bf16)accs[jt][it][v];
      *(bf16x8*)(eb + (size_t)i * NPIX + jb) = o;
    }
  }
}

// ---------------- Stage 3: per-row top-k threshold + masked softmax --------
// One block per row. Row cached in LDS; k-th largest value found by binary
// search on monotone float->uint keys; attn written in-place as bf16.
__global__ __launch_bounds__(256) void k_topk_softmax(__bf16* __restrict__ e,
                                                      const int* __restrict__ topk) {
  __shared__ float    vals[NPIX];
  __shared__ unsigned keys[NPIX];
  __shared__ float    red[256];
  __shared__ unsigned s_cnt;
  __shared__ float    s_sum;
  const int tid = threadIdx.x;
  __bf16* erow = e + (size_t)blockIdx.x * NPIX;

  for (int j = tid; j < NPIX; j += 256) {
    float f = (float)erow[j];
    vals[j] = f;
    unsigned u = __float_as_uint(f);
    keys[j] = (u & 0x80000000u) ? ~u : (u | 0x80000000u);
  }
  __syncthreads();

  const unsigned kk = (unsigned)(NPIX / topk[0]);  // k_keep = N/topk
  unsigned lo = 0u, hi = 0xFFFFFFFFu;              // find k-th largest key
  while (lo < hi) {
    const unsigned mid = lo + ((hi - lo) >> 1) + 1u;
    if (tid == 0) s_cnt = 0u;
    __syncthreads();
    unsigned c = 0;
    for (int j = tid; j < NPIX; j += 256) c += (keys[j] >= mid) ? 1u : 0u;
    if (c) atomicAdd(&s_cnt, c);
    __syncthreads();
    const unsigned tot = s_cnt;
    __syncthreads();
    if (tot >= kk) lo = mid; else hi = mid - 1u;
  }
  const unsigned thr = lo;

  float m = -3.4e38f;
  for (int j = tid; j < NPIX; j += 256) m = fmaxf(m, vals[j]);
  red[tid] = m;
  __syncthreads();
  for (int s = 128; s > 0; s >>= 1) {
    if (tid < s) red[tid] = fmaxf(red[tid], red[tid + s]);
    __syncthreads();
  }
  const float rowmax = red[0];

  if (tid == 0) s_sum = 0.0f;
  __syncthreads();
  float a = 0.0f;
  for (int j = tid; j < NPIX; j += 256)
    if (keys[j] >= thr) a += __expf(vals[j] - rowmax);
  atomicAdd(&s_sum, a);
  __syncthreads();
  const float inv = 1.0f / s_sum;

  for (int j = tid; j < NPIX; j += 256) {
    const float p = (keys[j] >= thr) ? __expf(vals[j] - rowmax) * inv : 0.0f;
    erow[j] = (__bf16)p;
  }
}

// ---------------- Stage 4: out[c][i] = sum_j attn[i][j]*v[c][j]; fuse ------
// 2x2 register-blocked D[m=i][n=c] tiles; epilogue gamma*acc + x with 32B f32
// vector load/store. attn (128 MB, streaming) is prefetched two K-panels
// ahead; v (16 MB) is L2-resident and needs no prefetch.
__global__ __launch_bounds__(256) void k_out(const __bf16* __restrict__ attn,
                                             const __bf16* __restrict__ vbf,
                                             const float* __restrict__ x,
                                             const float* __restrict__ gamma,
                                             float* __restrict__ out) {
  const int lane = threadIdx.x & 31;
  const int tile = blockIdx.x * 8 + (threadIdx.x >> 5);
  const int TI2 = NPIX / 32, TC2 = CH / 32;
  const int b = tile / (TI2 * TC2);
  const int t = tile % (TI2 * TC2);
  const int ti2 = t / TC2, tc2 = t % TC2;
  const __bf16* ab = attn + (size_t)b * NPIX * NPIX;
  const __bf16* vb = vbf + (size_t)b * CH * NPIX;
  const int r = lane & 15;
  v8f acc00 = {}, acc01 = {}, acc10 = {}, acc11 = {};
  for (int k = 0; k < NPIX; k += 32) {
    bf16x16 a0 = frag_a_bf16(ab, NPIX, ti2 * 32,      k, lane);  // M = i
    bf16x16 a1 = frag_a_bf16(ab, NPIX, ti2 * 32 + 16, k, lane);
    bf16x16 b0 = frag_b_bf16(vb, NPIX, tc2 * 32,      k, lane);  // N = c
    bf16x16 b1 = frag_b_bf16(vb, NPIX, tc2 * 32 + 16, k, lane);
    acc00 = WMMA_BF16(a0, b0, acc00);
    acc10 = WMMA_BF16(a1, b0, acc10);
    acc01 = WMMA_BF16(a0, b1, acc01);
    acc11 = WMMA_BF16(a1, b1, acc11);
    if (k + 64 < NPIX) {  // stream-ahead prefetch of the attn panel
      __builtin_prefetch(ab + (size_t)(ti2 * 32 + r) * NPIX + k + 64, 0, 1);
      __builtin_prefetch(ab + (size_t)(ti2 * 32 + 16 + r) * NPIX + k + 64, 0, 1);
    }
  }
  const int half = lane >> 4;
  const float g = gamma[0];
  const v8f accs[2][2] = {{acc00, acc01}, {acc10, acc11}};
#pragma unroll
  for (int it = 0; it < 2; ++it) {
#pragma unroll
    for (int ct = 0; ct < 2; ++ct) {
      const int c  = tc2 * 32 + ct * 16 + r;
      const int ib = ti2 * 32 + it * 16 + half * 8;
      const size_t base = ((size_t)b * CH + c) * NPIX + ib;
      f32x8 xv = *(const f32x8*)(x + base);
      f32x8 o;
#pragma unroll
      for (int v = 0; v < 8; ++v) o[v] = g * accs[it][ct][v] + xv[v];
      *(f32x8*)(out + base) = o;
    }
  }
}

// ---------------------------------------------------------------------------
extern "C" void kernel_launch(void* const* d_in, const int* in_sizes, int n_in,
                              void* d_out, int out_size, void* d_ws, size_t ws_size,
                              hipStream_t stream) {
  (void)in_sizes; (void)n_in; (void)out_size; (void)ws_size;
  const float* x     = (const float*)d_in[0];
  const float* Wq    = (const float*)d_in[1];
  const float* bq    = (const float*)d_in[2];
  const float* Wk    = (const float*)d_in[3];
  const float* bk    = (const float*)d_in[4];
  const float* Wv    = (const float*)d_in[5];
  const float* bv    = (const float*)d_in[6];
  const float* gamma = (const float*)d_in[7];
  const int*   topk  = (const int*)d_in[8];
  float* out = (float*)d_out;

  char* ws = (char*)d_ws;
  size_t off = 0;
  auto alloc = [&](size_t bytes) -> void* {
    void* p = (void*)(ws + off);
    off += (bytes + 255) & ~(size_t)255;
    return p;
  };
  __bf16* xT  = (__bf16*)alloc((size_t)BATCH * NPIX * CH * 2);
  __bf16* qT  = (__bf16*)alloc((size_t)BATCH * NPIX * KDIM * 2);
  __bf16* kT  = (__bf16*)alloc((size_t)BATCH * NPIX * KDIM * 2);
  __bf16* vbf = (__bf16*)alloc((size_t)BATCH * CH * NPIX * 2);
  __bf16* e   = (__bf16*)alloc((size_t)BATCH * NPIX * NPIX * 2);

  // Stage 0: transpose + f32->bf16
  k_transpose<<<dim3(NPIX / 32, CH / 32, BATCH), dim3(32, 32), 0, stream>>>(x, xT);

  // Stage 1: projections (8 waves/block, one 16x16 WMMA tile per wave)
  k_proj_qk<<<(BATCH * (KDIM / 16) * (NPIX / 16)) / 8, 256, 0, stream>>>(Wq, bq, xT, qT);
  k_proj_qk<<<(BATCH * (KDIM / 16) * (NPIX / 16)) / 8, 256, 0, stream>>>(Wk, bk, xT, kT);
  k_proj_v <<<(BATCH * (NPIX / 16) * (CH / 16)) / 8, 256, 0, stream>>>(Wv, bv, xT, vbf);

  // Stage 2: energy (bf16, stays in 192MB L2), 32x32 region per wave
  k_energy<<<(BATCH * (NPIX / 32) * (NPIX / 32)) / 8, 256, 0, stream>>>(qT, kT, e);

  // Stage 3: per-row top-k threshold + masked softmax (in-place)
  k_topk_softmax<<<BATCH * NPIX, 256, 0, stream>>>(e, topk);

  // Stage 4: out = gamma * (v @ attn^T) + x, 32x32 region per wave
  k_out<<<(BATCH * (NPIX / 32) * (CH / 32)) / 8, 256, 0, stream>>>(e, vbf, x, gamma, out);
}
